// Model_13666585936060
// MI455X (gfx1250) — matrix operand order repaired and checked by reading, not compiled, not run on hardware
//
#include <hip/hip_runtime.h>
#include <math.h>

typedef unsigned short ushort_t;
typedef __attribute__((ext_vector_type(16))) __bf16 v16bf;
typedef __attribute__((ext_vector_type(8)))  float  v8f;
typedef unsigned int v4u __attribute__((ext_vector_type(4)));
typedef int v4i __attribute__((ext_vector_type(4)));
typedef int v8i __attribute__((ext_vector_type(8)));

// ---- plain 16B vector (avoid HIP uint4 ctor issues with bit_cast) ----
struct __align__(16) Q4 { unsigned x, y, z, w; };
struct __align__(16) Q4x2 { Q4 a, b; };

static __device__ __forceinline__ ushort_t f2bf(float f) {
    unsigned u = __float_as_uint(f);
    u += 0x7FFFu + ((u >> 16) & 1u);          // round-to-nearest-even
    return (ushort_t)(u >> 16);
}

// A-fragment (16x32 bf16, ISA 7.12.2 layout): lanes0-15 K=[0..7,16..23], lanes16-31 K=[8..15,24..31]
static __device__ __forceinline__ v16bf load_frag_a(const ushort_t* p, int half) {
    Q4x2 t;
    t.a = *(const Q4*)(p + half * 8);
    t.b = *(const Q4*)(p + 16 + half * 8);
    return __builtin_bit_cast(v16bf, t);
}
// B-fragment (32x16 bf16): lanes0-15 K=[0..7,16..23], lanes16-31 K=[8..15,24..31] (matches A)
static __device__ __forceinline__ v16bf load_frag_b(const ushort_t* p, int half) {
    Q4x2 t;
    t.a = *(const Q4*)(p + half * 8);
    t.b = *(const Q4*)(p + 16 + half * 8);
    return __builtin_bit_cast(v16bf, t);
}

// ---- TDM: 2D bf16 tile global->LDS, with LDS padding 4 dwords per 128 dwords ----
static __device__ __forceinline__ void tdm_load_tile_bf16(unsigned lds_byte_off, const void* gp,
                                                          unsigned td0, unsigned td1,
                                                          unsigned long long stride0,
                                                          unsigned tl0, unsigned tl1) {
    unsigned long long ga = (unsigned long long)gp;
    v4u g0;
    g0[0] = 1u;                                               // count=1 valid user descriptor
    g0[1] = lds_byte_off;                                     // lds_addr
    g0[2] = (unsigned)(ga & 0xFFFFFFFFull);                   // global_addr[31:0]
    g0[3] = (unsigned)((ga >> 32) & 0x1FFFFFFull) | (2u << 30); // global_addr[56:32] | type=2
    v8i g1;
    g1[0] = (int)((1u << 16)       // data_size = 2 bytes
                | (1u << 20)       // pad_enable
                | (6u << 22)       // pad_interval: every 128 dwords (512B row)
                | (3u << 25));     // pad_amount: 4 dwords (16B -> stride 264 elems)
    g1[1] = (int)((td0 & 0xFFFFu) << 16);                     // tensor_dim0[15:0]
    g1[2] = (int)((td0 >> 16) | ((td1 & 0xFFFFu) << 16));     // tensor_dim0[31:16], tensor_dim1[15:0]
    g1[3] = (int)((td1 >> 16) | (tl0 << 16));                 // tensor_dim1[31:16], tile_dim0
    g1[4] = (int)(tl1 & 0xFFFFu);                             // tile_dim1 (tile_dim2=0)
    g1[5] = (int)(stride0 & 0xFFFFFFFFull);                   // tensor_dim0_stride[31:0]
    g1[6] = (int)((stride0 >> 32) & 0xFFFFull);               // stride[47:32]
    g1[7] = 0;
    v4i z4 = {0, 0, 0, 0};
    v8i z8 = {0, 0, 0, 0, 0, 0, 0, 0};
    // 6-arg variant (clang-23 / therock-10.0 headers): (g0, g1, g2, g3, extra, cpol)
    __builtin_amdgcn_tensor_load_to_lds(g0, g1, z4, z4, z8, 0);
}

// =====================================================================
// Kernel T: tiled transpose + f32->bf16 convert  (dst[c][r] = bf16(src[r][c]))
// =====================================================================
__global__ void transpose_to_bf16(const float* __restrict__ src, ushort_t* __restrict__ dst,
                                  int R, int C) {
    __shared__ float tile[32][33];
    int c0 = blockIdx.x * 32, r0 = blockIdx.y * 32;
    for (int i = threadIdx.y; i < 32; i += 8)
        tile[i][threadIdx.x] = src[(size_t)(r0 + i) * C + c0 + threadIdx.x];
    __syncthreads();
    for (int i = threadIdx.y; i < 32; i += 8)
        dst[(size_t)(c0 + i) * R + r0 + threadIdx.x] = f2bf(tile[threadIdx.x][i]);
}

// =====================================================================
// Kernel A: Xp[t*8+b][j] = embed[ids[b][t]] . W_xh[:,j]   (fp32), also zero sync flags
// =====================================================================
__global__ void embed_xh(const int* __restrict__ x_ids, const float* __restrict__ embed,
                         const float* __restrict__ W_xh, float* __restrict__ Xp,
                         int* __restrict__ flags) {
    __shared__ float xs[8][256];
    const int t = blockIdx.x;
    const int tid = threadIdx.x;
    if (blockIdx.x == 0 && tid < 256) flags[tid] = 0;     // reset per-step sync flags each call
    for (int idx = tid; idx < 2048; idx += 512) {
        int b = idx >> 8, k = idx & 255;
        int id = x_ids[b * 256 + t];
        xs[b][k] = embed[(size_t)id * 256 + k];
    }
    __syncthreads();
    const int j = tid;                                     // 0..511
    float acc[8] = {0, 0, 0, 0, 0, 0, 0, 0};
    for (int k = 0; k < 256; ++k) {
        float wk = W_xh[k * 512 + j];
#pragma unroll
        for (int b = 0; b < 8; ++b) acc[b] += xs[b][k] * wk;
    }
#pragma unroll
    for (int b = 0; b < 8; ++b) Xp[((size_t)t * 8 + b) * 512 + j] = acc[b];
}

// =====================================================================
// Kernel B: recurrence. 2 WGs x 512 threads; each WG owns 256 output cols,
// keeps its transposed W_hh slice (bf16) LDS-resident for all 256 steps.
// h exchanged through L2 with release/acquire flag per step.
// =====================================================================
__global__ void rnn_rec(const float* __restrict__ Xp, const float* __restrict__ b_h,
                        const ushort_t* __restrict__ WtHH, ushort_t* __restrict__ hs_bf,
                        ushort_t* __restrict__ hx, int* __restrict__ flags) {
    __shared__ __align__(16) ushort_t wsl[256 * 520];      // W_hh^T slice, padded stride 520
    __shared__ __align__(16) ushort_t hbuf[16 * 520];      // h (16x512 bf16), rows 8..15 stay zero
    const int tid  = threadIdx.x;
    const int wg   = blockIdx.x;                           // 0 or 1
    const int w    = tid >> 5;
    const int lane = tid & 31;
    const int m    = lane & 15;
    const int half = lane >> 4;

    // load weight slice: rows [wg*256, wg*256+256) of WtHH[512][512] (contiguous 256KB)
    for (int idx = tid; idx < 256 * 64; idx += 512) {
        int row = idx >> 6, q = idx & 63;
        ((Q4*)(wsl + row * 520))[q] = ((const Q4*)(WtHH + (size_t)(wg * 256 + row) * 512))[q];
    }
    for (int idx = tid; idx < (16 * 520) / 8; idx += 512)
        ((Q4*)hbuf)[idx] = Q4{0, 0, 0, 0};
    __syncthreads();

    const int j = wg * 256 + w * 16 + m;                   // this lane's output column (half==0)
    const float bh = b_h[j];

    for (int t = 0; t < 256; ++t) {
        v8f acc = {0, 0, 0, 0, 0, 0, 0, 0};
#pragma unroll
        for (int kc = 0; kc < 16; ++kc) {
            v16bf a  = load_frag_a(hbuf + m * 520 + kc * 32, half);
            v16bf bm = load_frag_b(wsl + (w * 16 + m) * 520 + kc * 32, half);
            acc = __builtin_amdgcn_wmma_f32_16x16x32_bf16(false, a, false, bm,
                                                          (short)0, acc, false, false);
        }
        if (half == 0) {                                   // lanes 0-15 hold batches 0..7
#pragma unroll
            for (int r = 0; r < 8; ++r) {
                float v = acc[r] + Xp[((size_t)t * 8 + r) * 512 + j] + bh;
                ushort_t hb = f2bf(tanhf(v));
                hs_bf[((size_t)r * 256 + t) * 512 + j] = hb;   // row = b*T + t
                hx[r * 512 + j] = hb;
            }
        }
        if (t == 255) break;
        __threadfence();                                   // publish hx (agent scope)
        __syncthreads();
        if (tid == 0) {
            __hip_atomic_fetch_add(&flags[t], 1, __ATOMIC_RELEASE, __HIP_MEMORY_SCOPE_AGENT);
            while (__hip_atomic_load(&flags[t], __ATOMIC_ACQUIRE, __HIP_MEMORY_SCOPE_AGENT) < 2)
                __builtin_amdgcn_s_sleep(1);
        }
        __syncthreads();
        __threadfence();                                   // invalidate stale lines before reread
        {   // refresh full h (rows 0..7) from exchange buffer: 8KB, one Q4 per thread
            int row = tid >> 6, q = tid & 63;
            ((Q4*)(hbuf + row * 520))[q] = ((const Q4*)hx)[tid];
        }
        __syncthreads();
    }
}

// =====================================================================
// Kernel C: Y[bt][v] = hs[bt][:] . W_hy[:,v] + b_y[v]
// 128x128 tile per WG, K=512 in two 256-chunks loaded by TDM into LDS.
// =====================================================================
__global__ void out_gemm(const ushort_t* __restrict__ hs_bf, const ushort_t* __restrict__ WtHy,
                         const float* __restrict__ b_y, float* __restrict__ Y) {
    __shared__ __align__(16) ushort_t As[128 * 264];       // hs tile (padded by TDM)
    __shared__ __align__(16) ushort_t Bs[128 * 264];       // W_hy^T tile (padded by TDM)
    const int tid  = threadIdx.x;
    const int w    = tid >> 5;
    const int lane = tid & 31;
    const int m    = lane & 15;
    const int half = lane >> 4;
    const int wr   = w >> 2, wc = w & 3;
    const int n0   = blockIdx.x * 128;
    const int m0   = blockIdx.y * 128;

    v8f acc00 = {0,0,0,0,0,0,0,0}, acc01 = acc00, acc10 = acc00, acc11 = acc00;

    for (int ko = 0; ko < 2; ++ko) {
        __syncthreads();                                   // prior compute done before overwrite
        if (tid < 32) {
            tdm_load_tile_bf16((unsigned)(unsigned long long)(void*)As,
                               hs_bf + ((size_t)m0 * 512 + ko * 256),
                               256, 128, 512ull, 256, 128);
            tdm_load_tile_bf16((unsigned)(unsigned long long)(void*)Bs,
                               WtHy + ((size_t)n0 * 512 + ko * 256),
                               256, 128, 512ull, 256, 128);
            __builtin_amdgcn_s_wait_tensorcnt(0);
        }
        __syncthreads();
#pragma unroll
        for (int kc = 0; kc < 8; ++kc) {
            const ushort_t* pa0 = As + (wr * 32 + m) * 264 + kc * 32;
            const ushort_t* pb0 = Bs + (wc * 32 + m) * 264 + kc * 32;
            v16bf a0 = load_frag_a(pa0, half);
            v16bf a1 = load_frag_a(pa0 + 16 * 264, half);
            v16bf b0 = load_frag_b(pb0, half);
            v16bf b1 = load_frag_b(pb0 + 16 * 264, half);
            acc00 = __builtin_amdgcn_wmma_f32_16x16x32_bf16(false, a0, false, b0, (short)0, acc00, false, false);
            acc01 = __builtin_amdgcn_wmma_f32_16x16x32_bf16(false, a0, false, b1, (short)0, acc01, false, false);
            acc10 = __builtin_amdgcn_wmma_f32_16x16x32_bf16(false, a1, false, b0, (short)0, acc10, false, false);
            acc11 = __builtin_amdgcn_wmma_f32_16x16x32_bf16(false, a1, false, b1, (short)0, acc11, false, false);
        }
    }

    const int colA = n0 + wc * 32 + m;
    const int colB = colA + 16;
    const float by0 = b_y[colA], by1 = b_y[colB];
    const int row0 = m0 + wr * 32 + half * 8;              // D layout: VGPR r -> M = half*8 + r
#pragma unroll
    for (int r = 0; r < 8; ++r) {
        size_t ra = (size_t)(row0 + r) * 32000;
        Y[ra + colA] = acc00[r] + by0;
        Y[ra + colB] = acc01[r] + by1;
        size_t rb = (size_t)(row0 + 16 + r) * 32000;
        Y[rb + colA] = acc10[r] + by0;
        Y[rb + colB] = acc11[r] + by1;
    }
}

// =====================================================================
// Host-side launch
// =====================================================================
extern "C" void kernel_launch(void* const* d_in, const int* in_sizes, int n_in,
                              void* d_out, int out_size, void* d_ws, size_t ws_size,
                              hipStream_t stream) {
    (void)in_sizes; (void)n_in; (void)out_size; (void)ws_size;
    const int*   x_ids = (const int*)d_in[0];
    const float* embed = (const float*)d_in[1];
    const float* W_xh  = (const float*)d_in[2];
    const float* b_h   = (const float*)d_in[3];
    const float* W_hh  = (const float*)d_in[4];
    const float* W_hy  = (const float*)d_in[5];
    const float* b_y   = (const float*)d_in[6];
    float* Y = (float*)d_out;

    char* ws = (char*)d_ws;
    constexpr size_t OFF_XP    = 0;                        // 2048x512 f32  = 4 MB
    constexpr size_t OFF_WTHH  = OFF_XP   + 2048ull * 512 * 4;   // 512x512 bf16 = 512 KB
    constexpr size_t OFF_WTHY  = OFF_WTHH + 512ull * 512 * 2;    // 32000x512 bf16 = 32.75 MB
    constexpr size_t OFF_HS    = OFF_WTHY + 32000ull * 512 * 2;  // 2048x512 bf16 = 2 MB
    constexpr size_t OFF_HX    = OFF_HS   + 2048ull * 512 * 2;   // 8x512 bf16
    constexpr size_t OFF_FLAGS = OFF_HX   + 8ull * 512 * 2 + 256;

    float*    Xp    = (float*)(ws + OFF_XP);
    ushort_t* WtHH  = (ushort_t*)(ws + OFF_WTHH);
    ushort_t* WtHy  = (ushort_t*)(ws + OFF_WTHY);
    ushort_t* hs_bf = (ushort_t*)(ws + OFF_HS);
    ushort_t* hx    = (ushort_t*)(ws + OFF_HX);
    int*      flags = (int*)(ws + OFF_FLAGS);

    transpose_to_bf16<<<dim3(16, 16),   dim3(32, 8), 0, stream>>>(W_hh, WtHH, 512, 512);
    transpose_to_bf16<<<dim3(1000, 16), dim3(32, 8), 0, stream>>>(W_hy, WtHy, 512, 32000);
    embed_xh<<<256, 512, 0, stream>>>(x_ids, embed, W_xh, Xp, flags);
    rnn_rec<<<2, 512, 0, stream>>>(Xp, b_h, WtHH, hs_bf, hx, flags);
    out_gemm<<<dim3(250, 16), 512, 0, stream>>>(hs_bf, WtHy, b_y, Y);
}
